// ColSentenceModel_26396869001815
// MI455X (gfx1250) — compile-verified, matrix operand order checked by reading, not run on hardware
//
#include <hip/hip_runtime.h>
#include <hip/hip_bf16.h>

// ---- Problem constants (from reference) ----
// query_h: [128, 32, 256] f32   doc_h: [128, 128, 256] f32
// W: [256, 128] f32             b: [128] f32
// out logits: [128, 128] f32
#define BQ 128
#define LQ 32
#define BD 128
#define LD 128
#define H  256
#define D  128

typedef __attribute__((ext_vector_type(16))) _Float16 v16h;
typedef __attribute__((ext_vector_type(8)))  _Float16 v8h;
typedef __attribute__((ext_vector_type(8)))  float    v8f;

union Frag {
    v16h f;
    v8h  h[2];
};

// ------------------------------------------------------------------
// Phase 1: E = l2norm(X @ W + b), written as f16.
// One token per 128-thread block; thread j owns output dim j.
// x-row (256 f32) staged in LDS; W column access is coalesced.
// ------------------------------------------------------------------
__global__ __launch_bounds__(128) void project_norm_kernel(
    const float* __restrict__ X, const float* __restrict__ W,
    const float* __restrict__ b, _Float16* __restrict__ out)
{
    __shared__ float xs[H];
    __shared__ float red[D];
    const int j = threadIdx.x;          // 0..127 output dim
    const int t = blockIdx.x;           // token index

    xs[j]       = X[(size_t)t * H + j];
    xs[j + 128] = X[(size_t)t * H + 128 + j];
    __syncthreads();

    float acc = b[j];
#pragma unroll 8
    for (int k = 0; k < H; ++k)
        acc = fmaf(xs[k], W[k * D + j], acc);

    red[j] = acc * acc;
    __syncthreads();
#pragma unroll
    for (int s = 64; s > 0; s >>= 1) {
        if (j < s) red[j] += red[j + s];
        __syncthreads();
    }
    const float inv = 1.0f / fmaxf(sqrtf(red[0]), 1e-12f);
    out[(size_t)t * D + j] = (_Float16)(acc * inv);
}

// ------------------------------------------------------------------
// Phase 2: max-sim. Block = 8 waves, one block per query batch qb.
// Each wave handles doc batches db = wave, wave+8, ...
// Per (qb,db): S = q[qb] (32x128) x d[db]^T (128x128), rowmax over
// doc tokens, sum over query tokens -> logits[qb][db].
// A fragments: 16x32 f16, lane%16 = M row, lane/16 selects K-half,
// two contiguous 16B chunks per lane (K = hs*8.. and 16+hs*8..).
// B loaded with the symmetric pattern from row-major doc tokens
// (lane%16 = N doc token), giving B = d^T for D = A*B + C.
// ------------------------------------------------------------------
__global__ __launch_bounds__(256) void maxsim_kernel(
    const _Float16* __restrict__ qe,   // [BQ*LQ, D] f16 normalized
    const _Float16* __restrict__ de,   // [BD*LD, D] f16 normalized
    float* __restrict__ logits)        // [BQ, BD]
{
    const int qb   = blockIdx.x;
    const int lane = threadIdx.x & 31;
    const int wave = threadIdx.x >> 5;   // 0..7
    const int row  = lane & 15;
    const int hs   = lane >> 4;          // 0 or 1: K-half selector

    // Load the 32x128 query tile once: 2 row-blocks x 4 K-steps.
    const _Float16* qbase = qe + (size_t)qb * LQ * D;
    Frag A[2][4];
#pragma unroll
    for (int rb = 0; rb < 2; ++rb) {
#pragma unroll
        for (int ks = 0; ks < 4; ++ks) {
            const _Float16* p = qbase + (rb * 16 + row) * D + ks * 32 + hs * 8;
            A[rb][ks].h[0] = *(const v8h*)(p);        // K = ks*32 + hs*8 + 0..7
            A[rb][ks].h[1] = *(const v8h*)(p + 16);   // K = ks*32 + 16 + hs*8 + 0..7
        }
    }

    for (int db = wave; db < BD; db += 8) {
        const _Float16* dbase = de + (size_t)db * LD * D;
        // Prefetch the next doc tile this wave will touch (global_prefetch_b8).
        const _Float16* nbase = de + (size_t)((db + 8) & (BD - 1)) * LD * D;
        __builtin_prefetch(nbase + (size_t)row * 512, 0, 1);

        v8f rmax0, rmax1;
#pragma unroll
        for (int i = 0; i < 8; ++i) { rmax0[i] = -3.0e38f; rmax1[i] = -3.0e38f; }

        for (int cb = 0; cb < 8; ++cb) {           // 8 blocks of 16 doc tokens
            Frag B[4];
            const _Float16* bp = dbase + (cb * 16 + row) * D + hs * 8;
#pragma unroll
            for (int ks = 0; ks < 4; ++ks) {
                B[ks].h[0] = *(const v8h*)(bp + ks * 32);
                B[ks].h[1] = *(const v8h*)(bp + ks * 32 + 16);
            }
            v8f c0 = {};
            v8f c1 = {};
#pragma unroll
            for (int ks = 0; ks < 4; ++ks) {       // K = 128 in 4 steps of 32
                c0 = __builtin_amdgcn_wmma_f32_16x16x32_f16(
                        false, A[0][ks].f, false, B[ks].f, (short)0, c0, false, false);
                c1 = __builtin_amdgcn_wmma_f32_16x16x32_f16(
                        false, A[1][ks].f, false, B[ks].f, (short)0, c1, false, false);
            }
#pragma unroll
            for (int i = 0; i < 8; ++i) {
                rmax0[i] = fmaxf(rmax0[i], c0[i]);
                rmax1[i] = fmaxf(rmax1[i], c1[i]);
            }
        }

        // C layout: VGPR i holds row M=i (lanes 0-15) / M=i+8 (lanes 16-31),
        // N = lane%16. Rowmax = xor-max within 16-lane halves; then sum the
        // 32 row maxima (8 VGPRs x 2 tiles per lane-half + one cross-half add).
        float sum = 0.0f;
#pragma unroll
        for (int i = 0; i < 8; ++i) {
            float m0 = rmax0[i];
            float m1 = rmax1[i];
#pragma unroll
            for (int m = 1; m <= 8; m <<= 1) {
                m0 = fmaxf(m0, __shfl_xor(m0, m, 32));
                m1 = fmaxf(m1, __shfl_xor(m1, m, 32));
            }
            sum += m0 + m1;
        }
        sum += __shfl_xor(sum, 16, 32);
        if (lane == 0) logits[qb * BD + db] = sum;
    }
}

// ------------------------------------------------------------------
// Launch: project q (4096 tokens) and d (16384 tokens) into f16
// workspace (1 MB + 4 MB), then the WMMA max-sim over 128x128 pairs.
// ------------------------------------------------------------------
extern "C" void kernel_launch(void* const* d_in, const int* in_sizes, int n_in,
                              void* d_out, int out_size, void* d_ws, size_t ws_size,
                              hipStream_t stream) {
    const float* query_h = (const float*)d_in[0];   // [128,32,256]
    const float* doc_h   = (const float*)d_in[1];   // [128,128,256]
    const float* W       = (const float*)d_in[2];   // [256,128]
    const float* b       = (const float*)d_in[3];   // [128]
    float* logits        = (float*)d_out;           // [128,128]

    _Float16* qe = (_Float16*)d_ws;                 // 4096*128  f16 = 1 MB
    _Float16* de = qe + (size_t)BQ * LQ * D;        // 16384*128 f16 = 4 MB

    project_norm_kernel<<<BQ * LQ, 128, 0, stream>>>(query_h, W, b, qe);
    project_norm_kernel<<<BD * LD, 128, 0, stream>>>(doc_h,  W, b, de);
    maxsim_kernel<<<BQ, 256, 0, stream>>>(qe, de, logits);
}
